// PhaseFunctionedNetwork_66099546685778
// MI455X (gfx1250) — matrix operand order repaired and checked by reading, not compiled, not run on hardware
//
#include <hip/hip_runtime.h>
#include <hip/hip_fp16.h>

typedef __attribute__((ext_vector_type(16))) _Float16 v16h;
typedef __attribute__((ext_vector_type(8)))  float    v8f;

namespace {

constexpr int kSlices = 4;
constexpr int kHid    = 512;
constexpr int kInDim  = 342;
constexpr int kInPad  = 352;   // 11 * 32, K-padded for 16x16x32 WMMA
constexpr int kOutDim = 311;
constexpr int kOutPad = 320;   // 20 * 16, N-padded to tile size
constexpr int kBatch  = 1024;

union HalfVec {
  v16h   v;
  float4 f4[2];
};

// ---------------------------------------------------------------------------
// Per-row cubic coefficients, mapped to slice order.
// ---------------------------------------------------------------------------
__global__ void pfnn_coef_kernel(const float* __restrict__ x,
                                 float* __restrict__ coef) {
  int b = blockIdx.x * blockDim.x + threadIdx.x;
  if (b >= kBatch) return;
  float phase = x[b * (kInDim + 1) + kInDim];
  float ps  = (float)kSlices * phase;
  float mu  = ps - floorf(ps);
  int   i1  = ((int)ps) % kSlices;           // phase in [0,1) -> ps >= 0
  float mu2 = mu * mu, mu3 = mu2 * mu;
  float c0 = -0.5f * mu3 + 1.0f * mu2 - 0.5f * mu;
  float c1 =  1.5f * mu3 - 2.5f * mu2 + 1.0f;
  float c2 = -1.5f * mu3 + 2.0f * mu2 + 0.5f * mu;
  float c3 =  0.5f * mu3 - 0.5f * mu2;
  float c[4] = {c0, c1, c2, c3};
  float w[4];
#pragma unroll
  for (int j = 0; j < 4; ++j) w[(i1 + j + 3) & 3] = c[j];  // slice (i1-1+j)%4
  reinterpret_cast<float4*>(coef)[b] = make_float4(w[0], w[1], w[2], w[3]);
}

// ---------------------------------------------------------------------------
// f32 -> f16 convert with row/col zero padding.
// ---------------------------------------------------------------------------
__global__ void pfnn_cvt_kernel(const float* __restrict__ src, int src_ld,
                                _Float16* __restrict__ dst,
                                int rows_v, int cols_v, int rows_p, int cols_p) {
  int idx = blockIdx.x * blockDim.x + threadIdx.x;
  int total = rows_p * cols_p;
  if (idx >= total) return;
  int r = idx / cols_p;
  int c = idx - r * cols_p;
  float val = 0.0f;
  if (r < rows_v && c < cols_v) val = src[(long)r * src_ld + c];
  dst[idx] = (_Float16)val;
}

// ---------------------------------------------------------------------------
// One wave computes a 32(M) x 16(N) output tile (2 WMMA M-tiles, register-
// blocked so every B fragment feeds 2 WMMAs). The 8 waves of a block share
// one N-tile (identical B addresses -> WGP$ hits) across 8 M-groups.
//
// D = sum_s coef_s(row) * (A @ B_s^T + bias_s), optional ELU, f16/f32 out.
// ---------------------------------------------------------------------------
template <bool kElu, bool kOutHalf>
__global__ void pfnn_layer_kernel(const _Float16* __restrict__ A,
                                  const _Float16* __restrict__ B,
                                  const float* __restrict__ bias,
                                  const float* __restrict__ coef,
                                  _Float16* __restrict__ out_h,
                                  float* __restrict__ out_f,
                                  int M, int Npad, int Nvalid, int K,
                                  int ldA, int ldB, int ldOut) {
  const int wib     = threadIdx.x >> 5;      // wave in block: 0..7
  const int lane    = threadIdx.x & 31;
  const int mgroups = M >> 5;                // 32-row groups
  const int mblocks = mgroups >> 3;          // blocks along M
  const int mb = blockIdx.x % mblocks;
  const int nt = blockIdx.x / mblocks;
  const int mg = mb * 8 + wib;
  if (mg >= mgroups || nt >= (Npad >> 4)) return;  // wave-uniform guard
  const int m0 = mg << 5;                    // first of 32 rows
  const int n0 = nt << 4;
  const int hi = lane >> 4;                  // which half of the wave
  const int lo = lane & 15;

  // A fragments: lane holds rows m0+lo and m0+16+lo (CDNA5 16-bit A layout:
  // K 0..7 / 8..15 split across lane halves, +16 for the upper 8 halves).
  const _Float16* arow0 = A + (long)(m0 + lo) * ldA;
  const _Float16* arow1 = arow0 + (long)16 * ldA;
  // B fragment: lane holds output column n0+lo; same K split as A.
  const _Float16* brow = B + (long)(n0 + lo) * ldB;
  const long bslice = (long)Npad * ldB;

  v8f accA[kSlices] = {};   // rows m0 .. m0+15
  v8f accB[kSlices] = {};   // rows m0+16 .. m0+31

  for (int kb = 0; kb < K; kb += 32) {
    HalfVec a0, a1;
    a0.f4[0] = *reinterpret_cast<const float4*>(arow0 + kb + hi * 8);
    a0.f4[1] = *reinterpret_cast<const float4*>(arow0 + kb + 16 + hi * 8);
    a1.f4[0] = *reinterpret_cast<const float4*>(arow1 + kb + hi * 8);
    a1.f4[1] = *reinterpret_cast<const float4*>(arow1 + kb + 16 + hi * 8);

    HalfVec bv[kSlices];
#pragma unroll
    for (int s = 0; s < kSlices; ++s) {
      const float4* p =
          reinterpret_cast<const float4*>(brow + s * bslice + kb + hi * 8);
      bv[s].f4[0] = p[0];
      bv[s].f4[1] = p[2];
    }
#pragma unroll
    for (int s = 0; s < kSlices; ++s) {
      accA[s] = __builtin_amdgcn_wmma_f32_16x16x32_f16(
          false, a0.v, false, bv[s].v, (short)0, accA[s], false, false);
      accB[s] = __builtin_amdgcn_wmma_f32_16x16x32_f16(
          false, a1.v, false, bv[s].v, (short)0, accB[s], false, false);
    }
  }

  // Epilogue: per-row interpolation of slice results + biases.
  const int col = n0 + lo;
  float bv0 = 0.f, bv1 = 0.f, bv2 = 0.f, bv3 = 0.f;
  if (col < Nvalid) {
    bv0 = bias[0 * Nvalid + col];
    bv1 = bias[1 * Nvalid + col];
    bv2 = bias[2 * Nvalid + col];
    bv3 = bias[3 * Nvalid + col];
  }
#pragma unroll
  for (int t = 0; t < 2; ++t) {
    const v8f* acc = t == 0 ? accA : accB;
#pragma unroll
    for (int v = 0; v < 8; ++v) {
      int row = m0 + t * 16 + v + 8 * hi;    // CDNA5 32-bit C/D layout
      float4 c = *reinterpret_cast<const float4*>(coef + row * 4);
      float val = c.x * (acc[0][v] + bv0) + c.y * (acc[1][v] + bv1) +
                  c.z * (acc[2][v] + bv2) + c.w * (acc[3][v] + bv3);
      if (kElu) val = val > 0.f ? val : (__expf(val) - 1.f);
      if (kOutHalf) {
        out_h[(long)row * ldOut + col] = (_Float16)val;
      } else if (col < Nvalid) {
        out_f[(long)row * ldOut + col] = val;
      }
    }
  }
}

}  // namespace

extern "C" void kernel_launch(void* const* d_in, const int* in_sizes, int n_in,
                              void* d_out, int out_size, void* d_ws, size_t ws_size,
                              hipStream_t stream) {
  (void)in_sizes; (void)n_in; (void)out_size; (void)ws_size;
  const float* x  = (const float*)d_in[0];
  const float* W0 = (const float*)d_in[1];
  const float* W1 = (const float*)d_in[2];
  const float* W2 = (const float*)d_in[3];
  const float* b0 = (const float*)d_in[4];
  const float* b1 = (const float*)d_in[5];
  const float* b2 = (const float*)d_in[6];
  float* out = (float*)d_out;

  // Workspace carve-out (256B aligned slabs), ~7.7 MB total.
  char* ws = (char*)d_ws;
  size_t off = 0;
  auto carve = [&](size_t bytes) {
    void* p = ws + off;
    off = (off + bytes + 255) & ~(size_t)255;
    return p;
  };
  float*    coef = (float*)   carve((size_t)kBatch * 4 * sizeof(float));
  _Float16* Xh   = (_Float16*)carve((size_t)kBatch * kInPad * sizeof(_Float16));
  _Float16* W0h  = (_Float16*)carve((size_t)kSlices * kHid * kInPad * sizeof(_Float16));
  _Float16* W1h  = (_Float16*)carve((size_t)kSlices * kHid * kHid * sizeof(_Float16));
  _Float16* W2h  = (_Float16*)carve((size_t)kSlices * kOutPad * kHid * sizeof(_Float16));
  _Float16* H1h  = (_Float16*)carve((size_t)kBatch * kHid * sizeof(_Float16));
  _Float16* H2h  = (_Float16*)carve((size_t)kBatch * kHid * sizeof(_Float16));

  // --- Prep: coefficients + f16 conversion with padding -------------------
  pfnn_coef_kernel<<<(kBatch + 255) / 256, 256, 0, stream>>>(x, coef);

  {
    int total = kBatch * kInPad;
    pfnn_cvt_kernel<<<(total + 255) / 256, 256, 0, stream>>>(
        x, kInDim + 1, Xh, kBatch, kInDim, kBatch, kInPad);
  }
  {
    int total = kSlices * kHid * kInPad;
    pfnn_cvt_kernel<<<(total + 255) / 256, 256, 0, stream>>>(
        W0, kInDim, W0h, kSlices * kHid, kInDim, kSlices * kHid, kInPad);
  }
  {
    int total = kSlices * kHid * kHid;
    pfnn_cvt_kernel<<<(total + 255) / 256, 256, 0, stream>>>(
        W1, kHid, W1h, kSlices * kHid, kHid, kSlices * kHid, kHid);
  }
  for (int s = 0; s < kSlices; ++s) {
    int total = kOutPad * kHid;
    pfnn_cvt_kernel<<<(total + 255) / 256, 256, 0, stream>>>(
        W2 + (size_t)s * kOutDim * kHid, kHid,
        W2h + (size_t)s * kOutPad * kHid, kOutDim, kHid, kOutPad, kHid);
  }

  // --- 3 WMMA layers -------------------------------------------------------
  // Grid: (M/32/8) M-blocks x (Npad/16) N-tiles; 8 wave32 per block, each
  // wave owns a 32x16 tile, all waves in a block share the N-tile's B data.
  constexpr int kMBlocks = kBatch / 32 / 8;   // 4
  {
    int blocks = kMBlocks * (kHid / 16);      // 4 * 32 = 128
    pfnn_layer_kernel<true, true><<<blocks, 256, 0, stream>>>(
        Xh, W0h, b0, coef, H1h, nullptr,
        kBatch, kHid, kHid, kInPad, kInPad, kInPad, kHid);
    pfnn_layer_kernel<true, true><<<blocks, 256, 0, stream>>>(
        H1h, W1h, b1, coef, H2h, nullptr,
        kBatch, kHid, kHid, kHid, kHid, kHid, kHid);
  }
  {
    int blocks = kMBlocks * (kOutPad / 16);   // 4 * 20 = 80
    pfnn_layer_kernel<false, false><<<blocks, 256, 0, stream>>>(
        H2h, W2h, b2, coef, nullptr, out,
        kBatch, kOutPad, kOutDim, kHid, kHid, kHid, kOutDim);
  }
}
